// DisattentionFormerV2_3607772529035
// MI455X (gfx1250) — matrix operand order repaired and checked by reading, not compiled, not run on hardware
//
#include <hip/hip_runtime.h>
#include <hip/hip_bf16.h>

// ---------------------------------------------------------------------------
// Problem constants (match reference)
// ---------------------------------------------------------------------------
#define Bc   4
#define Sc   2048
#define Dc   1024
#define CHc  64
#define Nc   32            // S / CH
#define BCc  256           // B * CH (rows per chunk in the scan)
#define EPSc 1.1920929e-07f
#define MEM_LR    0.1f
#define MEM_MOM   0.9f
#define MEM_DECAY 0.01f

typedef __bf16 bf16_t;
typedef bf16_t v16bf __attribute__((ext_vector_type(16)));
typedef bf16_t v8bf  __attribute__((ext_vector_type(8)));
typedef float  v8f   __attribute__((ext_vector_type(8)));

#define BKt 32

// ---- CDNA5 async global->LDS copy (ASYNCcnt path), feature-guarded ----
// Probe result: builtin expects (int4 addrspace(1)*, <lds ptr>, imm, imm).
#if defined(__AMDGCN__) && __has_builtin(__builtin_amdgcn_global_load_async_to_lds_b128)
#define ASYNC_LDS 1
typedef int v4i_gcc __attribute__((vector_size(4 * sizeof(int))));
typedef __attribute__((address_space(1))) v4i_gcc gbl_v4i;
typedef __attribute__((address_space(3))) v4i_gcc lds_v4i;
#define ASYNC_CP128(srcp, dstp)                                                  \
    __builtin_amdgcn_global_load_async_to_lds_b128((gbl_v4i*)(srcp),             \
                                                   (lds_v4i*)(dstp), 0, 0)
#else
#define ASYNC_LDS 0
#endif

__device__ inline void wait_async0()
{
#if defined(__AMDGCN__)
#if __has_builtin(__builtin_amdgcn_s_wait_asynccnt)
    __builtin_amdgcn_s_wait_asynccnt(0);
#else
    asm volatile("s_wait_asynccnt 0x0" ::: "memory");
#endif
#endif
}

// ---------------------------------------------------------------------------
// WMMA GEMM:  C[M,N] = epilogue( opA(A) @ opB(B) ),  A/B stored as bf16.
//   TBM/TBN : block tile (128x128 for big projections, 64x64 for scan gemms)
//   TA==0: A stored [M,K].   TA==1: A stored [K,M] (computes A^T @ ...)
//   TB==1: B stored [N,K] (weight, B^T).  TB==0: B stored [K,N].
//   ACT: 0 none | 1 SiLU | 2 loss-grad: (acc - V[m,n]) * escale
//   OUTBF: 1 -> C is bf16, 0 -> C is fp32.
// 256 threads = 8 waves (4 along M, 2 along N); fp32 accumulate in WMMA.
// All dims are multiples of the tile sizes for this problem (no guards).
// Contiguous operand tiles are moved with GLOBAL_LOAD_ASYNC_TO_LDS_B128
// (ASYNCcnt) when available; transposed tiles use batched register staging.
// ---------------------------------------------------------------------------
template <int TBM, int TBN, int TA, int TB, int ACT, int OUTBF>
__global__ __launch_bounds__(256)
void gemm_wmma(const bf16_t* __restrict__ A, const bf16_t* __restrict__ B,
               void* __restrict__ Cv, const bf16_t* __restrict__ Vb, float escale,
               int M, int N, int K,
               long long strideA, long long strideB, long long strideC)
{
    constexpr int FM = TBM / 64;   // fragments per wave along M (4 waves in M)
    constexpr int FN = TBN / 32;   // fragments per wave along N (2 waves in N)
    __shared__ __attribute__((aligned(16))) bf16_t sA[TBM][BKt];
    __shared__ __attribute__((aligned(16))) bf16_t sB[TBN][BKt];

    A += (size_t)blockIdx.z * strideA;
    B += (size_t)blockIdx.z * strideB;

    const int tid   = threadIdx.x;
    const int lane  = tid & 31;
    const int wave  = tid >> 5;
    const int waveM = wave & 3;
    const int waveN = wave >> 2;
    const int half  = lane >> 4;
    const int l15   = lane & 15;
    const int blockM = blockIdx.y * TBM;
    const int blockN = blockIdx.x * TBN;

    v8f acc[FM][FN];
#pragma unroll
    for (int i = 0; i < FM; ++i)
#pragma unroll
        for (int j = 0; j < FN; ++j)
#pragma unroll
            for (int e = 0; e < 8; ++e) acc[i][j][e] = 0.0f;

    for (int k0 = 0; k0 < K; k0 += BKt) {
        // ================= fill sA[m][k] =================
        if constexpr (TA == 0) {
            constexpr int NCH = (TBM * 4) / 256;   // 16B chunks per thread
#if ASYNC_LDS
#pragma unroll
            for (int i = 0; i < NCH; ++i) {
                const int idx = tid + i * 256;
                const int row = idx >> 2, seg = idx & 3;
                ASYNC_CP128(&A[(size_t)(blockM + row) * K + k0 + seg * 8],
                            &sA[row][seg * 8]);
            }
#else
            v8bf tmpA[NCH];
#pragma unroll
            for (int i = 0; i < NCH; ++i) {         // issue all loads first
                const int idx = tid + i * 256;
                tmpA[i] = *(const v8bf*)&A[(size_t)(blockM + (idx >> 2)) * K + k0 + (idx & 3) * 8];
            }
#pragma unroll
            for (int i = 0; i < NCH; ++i) {         // then store to LDS
                const int idx = tid + i * 256;
                *(v8bf*)&sA[idx >> 2][(idx & 3) * 8] = tmpA[i];
            }
#endif
        } else { // A stored [K,M]; transpose into LDS via registers
            constexpr int NTR = TBM / 64;
            v8bf tmpA[NTR];
#pragma unroll
            for (int i = 0; i < NTR; ++i) {
                const int idx = tid + i * 256;
                tmpA[i] = *(const v8bf*)&A[(size_t)(k0 + (idx & 31)) * M + blockM + (idx >> 5) * 8];
            }
            if (k0 + BKt < K)
                __builtin_prefetch(&A[(size_t)(k0 + BKt + (tid & 31)) * M + blockM], 0, 0);
#pragma unroll
            for (int i = 0; i < NTR; ++i) {
                const int idx = tid + i * 256;
                const int kr = idx & 31, seg = idx >> 5;
#pragma unroll
                for (int c = 0; c < 8; ++c) sA[seg * 8 + c][kr] = tmpA[i][c];
            }
        }
        // ================= fill sB[n][k] =================
        if constexpr (TB == 1) {
            constexpr int NCH = (TBN * 4) / 256;
#if ASYNC_LDS
#pragma unroll
            for (int i = 0; i < NCH; ++i) {
                const int idx = tid + i * 256;
                const int row = idx >> 2, seg = idx & 3;
                ASYNC_CP128(&B[(size_t)(blockN + row) * K + k0 + seg * 8],
                            &sB[row][seg * 8]);
            }
#else
            v8bf tmpB[NCH];
#pragma unroll
            for (int i = 0; i < NCH; ++i) {
                const int idx = tid + i * 256;
                tmpB[i] = *(const v8bf*)&B[(size_t)(blockN + (idx >> 2)) * K + k0 + (idx & 3) * 8];
            }
#pragma unroll
            for (int i = 0; i < NCH; ++i) {
                const int idx = tid + i * 256;
                *(v8bf*)&sB[idx >> 2][(idx & 3) * 8] = tmpB[i];
            }
#endif
        } else { // B stored [K,N]; transpose into LDS via registers
            constexpr int NTR = TBN / 64;
            v8bf tmpB[NTR];
#pragma unroll
            for (int i = 0; i < NTR; ++i) {
                const int idx = tid + i * 256;
                tmpB[i] = *(const v8bf*)&B[(size_t)(k0 + (idx & 31)) * N + blockN + (idx >> 5) * 8];
            }
            if (k0 + BKt < K)
                __builtin_prefetch(&B[(size_t)(k0 + BKt + (tid & 31)) * N + blockN], 0, 0);
#pragma unroll
            for (int i = 0; i < NTR; ++i) {
                const int idx = tid + i * 256;
                const int kr = idx & 31, seg = idx >> 5;
#pragma unroll
                for (int c = 0; c < 8; ++c) sB[seg * 8 + c][kr] = tmpB[i][c];
            }
        }
#if ASYNC_LDS
        if constexpr (TA == 0 || TB == 1) wait_async0();   // drain ASYNCcnt
#endif
        __syncthreads();

        // ---- A fragments (ISA wave32 16-bit A layout) ----
        v16bf afrag[FM];
#pragma unroll
        for (int i = 0; i < FM; ++i) {
            const int m  = waveM * (FM * 16) + i * 16 + l15;
            const int kb = half * 8;
            const v8bf lo = *(const v8bf*)&sA[m][kb];
            const v8bf hi = *(const v8bf*)&sA[m][kb + 16];
#pragma unroll
            for (int j = 0; j < 8; ++j) { afrag[i][j] = lo[j]; afrag[i][j + 8] = hi[j]; }
        }
        // ---- B fragments (lane = N column; same K order as A fragments) ----
        v16bf bfrag[FN];
#pragma unroll
        for (int jn = 0; jn < FN; ++jn) {
            const int n  = waveN * (FN * 16) + jn * 16 + l15;
            const int kb = half * 8;
            const v8bf lo = *(const v8bf*)&sB[n][kb];
            const v8bf hi = *(const v8bf*)&sB[n][kb + 16];
#pragma unroll
            for (int j = 0; j < 8; ++j) { bfrag[jn][j] = lo[j]; bfrag[jn][j + 8] = hi[j]; }
        }
#pragma unroll
        for (int i = 0; i < FM; ++i)
#pragma unroll
            for (int jn = 0; jn < FN; ++jn)
                acc[i][jn] = __builtin_amdgcn_wmma_f32_16x16x32_bf16(
                    false, afrag[i], false, bfrag[jn], (short)0, acc[i][jn], false, false);
        __syncthreads();
    }

    // ---- epilogue + store (C/D layout: lane=N col, VGPR v -> M = v+8*half) ----
#pragma unroll
    for (int i = 0; i < FM; ++i) {
        const int mbase = blockM + waveM * (FM * 16) + i * 16 + half * 8;
#pragma unroll
        for (int jn = 0; jn < FN; ++jn) {
            const int n = blockN + waveN * (FN * 16) + jn * 16 + l15;
#pragma unroll
            for (int v = 0; v < 8; ++v) {
                float val = acc[i][jn][v];
                if constexpr (ACT == 1) val = val / (1.0f + __expf(-val));      // SiLU
                if constexpr (ACT == 2)                                          // dL/dpred
                    val = (val - (float)Vb[(size_t)(mbase + v) * N + n]) * escale;
                const size_t cidx = (size_t)blockIdx.z * strideC + (size_t)(mbase + v) * N + n;
                if constexpr (OUTBF) ((bf16_t*)Cv)[cidx] = (bf16_t)val;
                else                 ((float*)Cv)[cidx]  = val;
            }
        }
    }
    (void)escale; (void)Vb;
}

template <int TBM, int TBN, int TA, int TB, int ACT, int OUTBF>
static inline void launch_gemm(const bf16_t* A, const bf16_t* B, void* C,
                               const bf16_t* Vb, float escale,
                               int M, int N, int K,
                               long long sA, long long sB, long long sC,
                               int batch, hipStream_t st)
{
    dim3 grid(N / TBN, M / TBM, batch);
    gemm_wmma<TBM, TBN, TA, TB, ACT, OUTBF><<<grid, 256, 0, st>>>(A, B, C, Vb, escale,
                                                                  M, N, K, sA, sB, sC);
}

// ---------------------------------------------------------------------------
// Elementwise / reduction helpers
// ---------------------------------------------------------------------------
__device__ inline float block_reduce_sum(float s)
{
    __shared__ float red[256];
    const int tid = threadIdx.x;
    red[tid] = s;
    __syncthreads();
    for (int st = 128; st > 0; st >>= 1) {
        if (tid < st) red[tid] += red[tid + st];
        __syncthreads();
    }
    return red[0];
}

__global__ __launch_bounds__(256) void convert_f2b_kernel(const float* __restrict__ x,
                                                          bf16_t* __restrict__ y, int n)
{
    const int i = blockIdx.x * 256 + threadIdx.x;
    if (i < n) y[i] = (bf16_t)x[i];
}

// y_bf16 = x * rsqrt(mean(x^2)+EPS) * g ; one block per row of D
__global__ __launch_bounds__(256) void rmsnorm_bf_kernel(const float* __restrict__ x,
                                                         const float* __restrict__ g,
                                                         bf16_t* __restrict__ y)
{
    const size_t row = blockIdx.x;
    const float* xr = x + row * Dc;
    bf16_t* yr = y + row * Dc;
    float s = 0.0f;
    for (int i = threadIdx.x; i < Dc; i += 256) { const float v = xr[i]; s += v * v; }
    const float tot = block_reduce_sum(s);
    const float scale = rsqrtf(tot / (float)Dc + EPSc);
    for (int i = threadIdx.x; i < Dc; i += 256) yr[i] = (bf16_t)(xr[i] * scale * g[i]);
}

// y_bf16 = l2norm(x_f32) per row of D
__global__ __launch_bounds__(256) void l2norm_bf_kernel(const float* __restrict__ x,
                                                        bf16_t* __restrict__ y)
{
    const size_t row = blockIdx.x;
    const float* xr = x + row * Dc;
    bf16_t* yr = y + row * Dc;
    float s = 0.0f;
    for (int i = threadIdx.x; i < Dc; i += 256) { const float v = xr[i]; s += v * v; }
    const float tot = block_reduce_sum(s);
    const float inv = 1.0f / fmaxf(sqrtf(tot), 1e-12f);
    for (int i = threadIdx.x; i < Dc; i += 256) yr[i] = (bf16_t)(xr[i] * inv);
}

// cm[b][n][d] = mean_c x[b][n*CH+c][d]
__global__ __launch_bounds__(256) void chunk_mean_kernel(const float* __restrict__ x,
                                                         float* __restrict__ cm)
{
    const size_t idx = (size_t)blockIdx.x * 256 + threadIdx.x;
    if (idx >= (size_t)Bc * Nc * Dc) return;
    const int d = idx % Dc;
    const int n = (idx / Dc) % Nc;
    const int b = idx / ((size_t)Dc * Nc);
    float s = 0.0f;
    const float* base = x + ((size_t)b * Sc + (size_t)n * CHc) * Dc + d;
    for (int c = 0; c < CHc; ++c) s += base[(size_t)c * Dc];
    cm[idx] = s * (1.0f / (float)CHc);
}

// per (b,n,gate): sum_f sigmoid(cm[b,n,:] @ Wg[f,:] + bg[f]) -> atomicAdd gacc
__global__ __launch_bounds__(256) void gate_partial_kernel(
    const float* __restrict__ cm,
    const float* __restrict__ Wgd, const float* __restrict__ bgd,
    const float* __restrict__ Wgl, const float* __restrict__ bgl,
    const float* __restrict__ Wgm, const float* __restrict__ bgm,
    float* __restrict__ gacc)
{
    const int bn   = blockIdx.x;       // b*N + n
    const int gate = blockIdx.y;       // 0=decay 1=lr 2=mom
    const float* W  = (gate == 0) ? Wgd : (gate == 1) ? Wgl : Wgm;
    const float* bb = (gate == 0) ? bgd : (gate == 1) ? bgl : bgm;
    const float* cmr = cm + (size_t)bn * Dc;
    float s = 0.0f;
    for (int f = threadIdx.x; f < Dc; f += 256) {
        float dot = bb[f];
        const float* wr = W + (size_t)f * Dc;
        for (int d = 0; d < Dc; ++d) dot += cmr[d] * wr[d];
        s += 1.0f / (1.0f + __expf(-dot));
    }
    const float tot = block_reduce_sum(s);
    if (threadIdx.x == 0) atomicAdd(&gacc[gate * Nc + (bn % Nc)], tot);
}

__global__ void finalize_gates_kernel(const float* __restrict__ gacc,
                                      float* __restrict__ gates)
{
    const int i = blockIdx.x * 64 + threadIdx.x;
    if (i >= 3 * Nc) return;
    const int g = i / Nc;
    const float scale = (g == 0) ? MEM_DECAY : (g == 1) ? MEM_LR : MEM_MOM;
    gates[i] = (gacc[i] / (float)(Bc * Dc)) * scale;
}

// dir 0: [B,S,D] -> [N, B*CH, D] chunk layout; dir 1: inverse (bf16)
__global__ __launch_bounds__(256) void reorder_bf_kernel(const bf16_t* __restrict__ src,
                                                         bf16_t* __restrict__ dst, int dir)
{
    const size_t idx = (size_t)blockIdx.x * 256 + threadIdx.x;
    const int d = idx % Dc;
    size_t r = idx / Dc;
    const int c = r % CHc; r /= CHc;
    const int n = r % Nc;
    const int b = r / Nc;
    const size_t bsd = ((size_t)b * Sc + (size_t)n * CHc + c) * Dc + d;
    const size_t nbc = (((size_t)n * Bc + b) * CHc + c) * Dc + d;
    if (dir == 0) dst[nbc] = src[bsd];
    else          dst[bsd] = src[nbc];
}

__global__ __launch_bounds__(256) void silu_f2b_kernel(const float* __restrict__ x,
                                                       bf16_t* __restrict__ y, int n)
{
    const int i = blockIdx.x * 256 + threadIdx.x;
    if (i < n) { const float v = x[i]; y[i] = (bf16_t)(v / (1.0f + __expf(-v))); }
}

// dz_bf16 = dh * silu'(z)
__global__ __launch_bounds__(256) void dz_f2b_kernel(const float* __restrict__ dh,
                                                     const float* __restrict__ z,
                                                     bf16_t* __restrict__ dz, int n)
{
    const int i = blockIdx.x * 256 + threadIdx.x;
    if (i < n) {
        const float zi = z[i];
        const float sg = 1.0f / (1.0f + __expf(-zi));
        dz[i] = (bf16_t)(dh[i] * sg * (1.0f + zi * (1.0f - sg)));
    }
}

// S = eta*S - theta*G ; W = (1-alpha)*W + S ; also refresh bf16 weight copy
__global__ __launch_bounds__(256) void update_kernel(float* __restrict__ W,
                                                     float* __restrict__ Sm,
                                                     const float* __restrict__ G,
                                                     const float* __restrict__ gates,
                                                     int chunk,
                                                     bf16_t* __restrict__ Wbf)
{
    const int i = blockIdx.x * 256 + threadIdx.x;
    if (i >= Dc * Dc) return;
    const float a  = gates[chunk];
    const float th = gates[Nc + chunk];
    const float et = gates[2 * Nc + chunk];
    const float s = et * Sm[i] - th * G[i];
    Sm[i] = s;
    const float w = (1.0f - a) * W[i] + s;
    W[i] = w;
    Wbf[i] = (bf16_t)w;
}

// ---------------------------------------------------------------------------
// Orchestration
// ---------------------------------------------------------------------------
extern "C" void kernel_launch(void* const* d_in, const int* in_sizes, int n_in,
                              void* d_out, int out_size, void* d_ws, size_t ws_size,
                              hipStream_t stream)
{
    const float* x     = (const float*)d_in[0];
    const float* Mmet  = (const float*)d_in[1];
    const float* Wk    = (const float*)d_in[2];
    const float* Wv    = (const float*)d_in[3];
    const float* Wq    = (const float*)d_in[4];
    const float* Wo    = (const float*)d_in[5];
    const float* Wgd   = (const float*)d_in[6];
    const float* bgd   = (const float*)d_in[7];
    const float* Wgl   = (const float*)d_in[8];
    const float* bgl   = (const float*)d_in[9];
    const float* Wgm   = (const float*)d_in[10];
    const float* bgm   = (const float*)d_in[11];
    const float* g_sto = (const float*)d_in[12];
    const float* g_ret = (const float*)d_in[13];
    const float* W0in  = (const float*)d_in[14];
    const float* W1in  = (const float*)d_in[15];
    float* out = (float*)d_out;

    // ---- workspace carve-up (256B-aligned) ----
    const size_t BIG = (size_t)Bc * Sc * Dc;          // 8,388,608 elems
    const size_t DD  = (size_t)Dc * Dc;
    const size_t EW  = (size_t)BCc * Dc;              // 262,144 per-chunk elems
    char* base = (char*)d_ws;
    size_t off = 0;
    auto alloc = [&](size_t bytes) -> void* {
        off = (off + 255) & ~(size_t)255;
        void* p = base + off; off += bytes; return p;
    };
    bf16_t* b0 = (bf16_t*)alloc(BIG * 2);   // xs -> q -> retc
    bf16_t* b1 = (bf16_t*)alloc(BIG * 2);   // xr -> kc -> ret
    bf16_t* b2 = (bf16_t*)alloc(BIG * 2);   // k  -> vc
    bf16_t* b3 = (bf16_t*)alloc(BIG * 2);   // v  -> qc
    float*  f0 = (float*)alloc(BIG * 4);    // fp32 scratch (k_raw / q_raw)
    bf16_t* Pbf = (bf16_t*)alloc((size_t)Bc * DD * 2);    // Mmet @ Wk^T (bf16)
    bf16_t* Mm_bf = (bf16_t*)alloc((size_t)Bc * DD * 2);
    bf16_t* Wk_bf = (bf16_t*)alloc(DD * 2);
    bf16_t* Wv_bf = (bf16_t*)alloc(DD * 2);
    bf16_t* Wq_bf = (bf16_t*)alloc(DD * 2);
    bf16_t* Wo_bf = (bf16_t*)alloc(DD * 2);
    float*  W0c = (float*)alloc(DD * 4);
    float*  W1c = (float*)alloc(DD * 4);
    float*  Sm0 = (float*)alloc(DD * 4);
    float*  Sm1 = (float*)alloc(DD * 4);
    float*  G0  = (float*)alloc(DD * 4);
    float*  G1  = (float*)alloc(DD * 4);
    bf16_t* W0bf = (bf16_t*)alloc(DD * 2);
    bf16_t* W1bf = (bf16_t*)alloc(DD * 2);
    float*  zk  = (float*)alloc(EW * 4);
    float*  dh1 = (float*)alloc(EW * 4);
    bf16_t* hk  = (bf16_t*)alloc(EW * 2);
    bf16_t* hq  = (bf16_t*)alloc(EW * 2);
    bf16_t* dpr = (bf16_t*)alloc(EW * 2);
    bf16_t* dzb = (bf16_t*)alloc(EW * 2);
    float*  cm  = (float*)alloc((size_t)Bc * Nc * Dc * 4);
    float*  gacc  = (float*)alloc(3 * Nc * 4);
    float*  gates = (float*)alloc(3 * Nc * 4);

    const long long sDD = (long long)DD, sSD = (long long)Sc * Dc;
    const int EWB = (int)((EW + 255) / 256);
    const int CVB = (int)((DD + 255) / 256);
    const float dscale = 2.0f / (float)(BCc * Dc);

    // ---- gates (tiny, fp32 scalar path) ----
    (void)hipMemsetAsync(gacc, 0, 3 * Nc * sizeof(float), stream);
    chunk_mean_kernel<<<(Bc * Nc * Dc + 255) / 256, 256, 0, stream>>>(x, cm);
    gate_partial_kernel<<<dim3(Bc * Nc, 3), 256, 0, stream>>>(cm, Wgd, bgd, Wgl, bgl,
                                                              Wgm, bgm, gacc);
    finalize_gates_kernel<<<(3 * Nc + 63) / 64, 64, 0, stream>>>(gacc, gates);

    // ---- one-time bf16 conversions of parameters ----
    convert_f2b_kernel<<<(int)((Bc * DD + 255) / 256), 256, 0, stream>>>(Mmet, Mm_bf, Bc * DD);
    convert_f2b_kernel<<<CVB, 256, 0, stream>>>(Wk, Wk_bf, DD);
    convert_f2b_kernel<<<CVB, 256, 0, stream>>>(Wv, Wv_bf, DD);
    convert_f2b_kernel<<<CVB, 256, 0, stream>>>(Wq, Wq_bf, DD);
    convert_f2b_kernel<<<CVB, 256, 0, stream>>>(Wo, Wo_bf, DD);
    convert_f2b_kernel<<<CVB, 256, 0, stream>>>(W0in, W0bf, DD);
    convert_f2b_kernel<<<CVB, 256, 0, stream>>>(W1in, W1bf, DD);
    (void)hipMemcpyAsync(W0c, W0in, DD * sizeof(float), hipMemcpyDeviceToDevice, stream);
    (void)hipMemcpyAsync(W1c, W1in, DD * sizeof(float), hipMemcpyDeviceToDevice, stream);
    (void)hipMemsetAsync(Sm0, 0, DD * sizeof(float), stream);
    (void)hipMemsetAsync(Sm1, 0, DD * sizeof(float), stream);

    // ---- norms (bf16 outputs) ----
    rmsnorm_bf_kernel<<<Bc * Sc, 256, 0, stream>>>(x, g_sto, b0);   // xs
    rmsnorm_bf_kernel<<<Bc * Sc, 256, 0, stream>>>(x, g_ret, b1);   // xr

    // ---- projections (WMMA, bf16 operands) ----
    // P[b] = Mmet[b] @ Wk^T  (fused metric: (xs@M)@Wk^T == xs@(M@Wk^T))
    launch_gemm<128, 128, 0, 1, 0, 1>(Mm_bf, Wk_bf, Pbf, nullptr, 0.0f,
                                      Dc, Dc, Dc, sDD, 0, sDD, Bc, stream);
    // k_raw = silu(xs @ P[b])  (P stored [K,N], fp32 out for l2norm)
    launch_gemm<128, 128, 0, 0, 1, 0>(b0, Pbf, f0, nullptr, 0.0f,
                                      Sc, Dc, Dc, sSD, sDD, sSD, Bc, stream);
    l2norm_bf_kernel<<<Bc * Sc, 256, 0, stream>>>(f0, b2);          // k (bf16)
    // v = silu(xs @ Wv^T)  (bf16 out directly)
    launch_gemm<128, 128, 0, 1, 1, 1>(b0, Wv_bf, b3, nullptr, 0.0f,
                                      Bc * Sc, Dc, Dc, 0, 0, 0, 1, stream);
    // q_raw = silu(xr @ Wq^T)
    launch_gemm<128, 128, 0, 1, 1, 0>(b1, Wq_bf, f0, nullptr, 0.0f,
                                      Bc * Sc, Dc, Dc, 0, 0, 0, 1, stream);
    l2norm_bf_kernel<<<Bc * Sc, 256, 0, stream>>>(f0, b0);          // q (xs dead)

    // ---- chunk layout [N, B*CH, D] ----
    const int RB = (int)((BIG + 255) / 256);
    reorder_bf_kernel<<<RB, 256, 0, stream>>>(b2, b1, 0);           // kc  (xr dead)
    reorder_bf_kernel<<<RB, 256, 0, stream>>>(b3, b2, 0);           // vc  (k dead)
    reorder_bf_kernel<<<RB, 256, 0, stream>>>(b0, b3, 0);           // qc  (v dead)
    bf16_t* kc = b1; bf16_t* vc = b2; bf16_t* qc = b3; bf16_t* retc = b0;

    // ---- sequential scan over 32 chunks (64x64 tiles for occupancy) ----
    for (int t = 0; t < Nc; ++t) {
        const bf16_t* kt = kc + (size_t)t * EW;
        const bf16_t* vt = vc + (size_t)t * EW;
        const bf16_t* qt = qc + (size_t)t * EW;
        bf16_t* rt = retc + (size_t)t * EW;

        // retrieve BEFORE update (causal): r = silu(q@W0^T) @ W1^T
        launch_gemm<64, 64, 0, 1, 1, 1>(qt, W0bf, hq, nullptr, 0.0f,
                                        BCc, Dc, Dc, 0, 0, 0, 1, stream);
        launch_gemm<64, 64, 0, 1, 0, 1>(hq, W1bf, rt, nullptr, 0.0f,
                                        BCc, Dc, Dc, 0, 0, 0, 1, stream);

        // surprise gradient of mean((silu(k@W0^T)@W1^T - v)^2)
        launch_gemm<64, 64, 0, 1, 0, 0>(kt, W0bf, zk, nullptr, 0.0f,
                                        BCc, Dc, Dc, 0, 0, 0, 1, stream);
        silu_f2b_kernel<<<EWB, 256, 0, stream>>>(zk, hk, (int)EW);
        // dpred fused into pred-gemm epilogue: (hk@W1^T - v) * 2/count
        launch_gemm<64, 64, 0, 1, 2, 1>(hk, W1bf, dpr, vt, dscale,
                                        BCc, Dc, Dc, 0, 0, 0, 1, stream);
        // G1 = dpred^T @ hk
        launch_gemm<64, 64, 1, 0, 0, 0>(dpr, hk, G1, nullptr, 0.0f,
                                        Dc, Dc, BCc, 0, 0, 0, 1, stream);
        // dh1 = dpred @ W1
        launch_gemm<64, 64, 0, 0, 0, 0>(dpr, W1bf, dh1, nullptr, 0.0f,
                                        BCc, Dc, Dc, 0, 0, 0, 1, stream);
        dz_f2b_kernel<<<EWB, 256, 0, stream>>>(dh1, zk, dzb, (int)EW);
        // G0 = dz^T @ k
        launch_gemm<64, 64, 1, 0, 0, 0>(dzb, kt, G0, nullptr, 0.0f,
                                        Dc, Dc, BCc, 0, 0, 0, 1, stream);

        // momentum + decay updates (fp32 masters, bf16 copies refreshed)
        update_kernel<<<CVB, 256, 0, stream>>>(W0c, Sm0, G0, gates, t, W0bf);
        update_kernel<<<CVB, 256, 0, stream>>>(W1c, Sm1, G1, gates, t, W1bf);
    }

    // ---- back to [B,S,D] and output projection (fp32 out) ----
    reorder_bf_kernel<<<RB, 256, 0, stream>>>(retc, b1, 1);         // ret (kc dead)
    launch_gemm<128, 128, 0, 1, 0, 0>(b1, Wo_bf, out, nullptr, 0.0f,
                                      Bc * Sc, Dc, Dc, 0, 0, 0, 1, stream);

    (void)in_sizes; (void)n_in; (void)out_size; (void)ws_size;
}
